// AttentionWithComplexRoPE_28166395527997
// MI455X (gfx1250) — matrix operand order repaired and checked by reading, not compiled, not run on hardware
//
#include <hip/hip_runtime.h>

// ---------------------------------------------------------------------------
// MI455X (gfx1250) attention-with-RoPE.
// Compute-bound (~69 GFLOP vs ~4us of HBM time at 23.3 TB/s) => everything
// through v_wmma_f32_16x16x32_bf16 (8x the FLOP/instr of the f32 WMMA path),
// f32 accumulate, f32 softmax via v_exp_f32 (log2-domain).
// GEMMs stage A/W tiles into LDS with the Tensor Data Mover (double-buffered,
// TENSORcnt-tracked), using TDM padding for a 272B LDS row pitch so fragment
// reads are bank-conflict-free.
// ---------------------------------------------------------------------------

typedef __attribute__((ext_vector_type(16))) __bf16 v16bf;
typedef __attribute__((ext_vector_type(8)))  __bf16 v8bf;
typedef __attribute__((ext_vector_type(8)))  float  v8f;
typedef __attribute__((ext_vector_type(4)))  unsigned int v4u;
typedef __attribute__((ext_vector_type(8)))  int v8i;
typedef __attribute__((ext_vector_type(4)))  int v4i;

#define SEQ_B   2
#define SEQ_S   2048
#define DMODEL  1024
#define NHEADS  16
#define HDIM    64
#define KC      128   // K-chunk staged per TDM transfer
#define LROW    136   // LDS row pitch in elements: 128 data + 8 pad (TDM pad)

static __device__ __forceinline__ v8bf ld8(const __bf16* p) {
  return *reinterpret_cast<const v8bf*>(p);
}
static __device__ __forceinline__ v16bf cat16(v8bf lo, v8bf hi) {
  return __builtin_shufflevector(lo, hi, 0,1,2,3,4,5,6,7,8,9,10,11,12,13,14,15);
}
static __device__ __forceinline__ v8f wmma_bf16(v16bf a, v16bf b, v8f c) {
  return __builtin_amdgcn_wmma_f32_16x16x32_bf16(false, a, false, b, (short)0, c,
                                                 false, false);
}

// ---- Tensor Data Mover: 2D bf16 tile (rows x KC) from global to LDS -------
// D# per CDNA5 ISA. Padding: every 64 DWORDs (=256B row)
// insert 4 DWORDs (16B) -> LDS pitch 272B = LROW elements.
static __device__ __forceinline__ void tdm_load_2d(unsigned lds_off,
                                                   const void* gptr,
                                                   int rows, int stride_elems) {
  unsigned long long ga = (unsigned long long)(size_t)gptr;
  v4u g0;
  g0[0] = 1u;                                  // count=1, user mode
  g0[1] = lds_off;                             // lds_addr (bytes)
  g0[2] = (unsigned)ga;                        // global_addr[31:0]
  g0[3] = (unsigned)(ga >> 32) | (2u << 30);   // global_addr[56:32] | type=2
  v8i g1;
  g1[0] = (int)((1u << 16)      // data_size = 2 bytes
              | (1u << 20)      // pad_enable
              | (5u << 22)      // pad_interval: 64 DWORDs
              | (3u << 25));    // pad_amount: 4 DWORDs
  g1[1] = (int)((unsigned)KC << 16);                    // tensor_dim0[15:0]
  g1[2] = (int)((unsigned)rows << 16);                  // tensor_dim1[15:0]
  g1[3] = (int)((unsigned)KC << 16);                    // tile_dim0
  g1[4] = rows;                                         // tile_dim1 (tile_dim2=0)
  g1[5] = stride_elems;                                 // tensor_dim0_stride lo
  g1[6] = 0;
  g1[7] = 0;
  v4i gz = {0, 0, 0, 0};
#if defined(__clang_major__) && __clang_major__ >= 23
  v8i gz8 = {0, 0, 0, 0, 0, 0, 0, 0};
  __builtin_amdgcn_tensor_load_to_lds(g0, g1, gz, gz, gz8, 0);
#else
  __builtin_amdgcn_tensor_load_to_lds(g0, g1, gz, gz, 0);
#endif
}

// -------------------------------- convert ----------------------------------
__global__ void k_f32_to_bf16(const float* __restrict__ src,
                              __bf16* __restrict__ dst, int n) {
  int i = blockIdx.x * blockDim.x + threadIdx.x;
  if (i < n) dst[i] = (__bf16)src[i];
}

// --------------------------------- GEMM ------------------------------------
// C[M,N] = A[M,K] (bf16) @ W[N,K]^T (bf16). Block = 4 waves = 64x64 C tile.
// TDM double-buffers the 64xKC A-tile and W-tile in LDS; waves build WMMA
// fragments from LDS (conflict-free thanks to the 272B padded pitch).
__global__ __launch_bounds__(128) void k_gemm_bf16(
    const __bf16* __restrict__ A, const __bf16* __restrict__ W,
    float* __restrict__ C, int M, int N, int K) {
  __shared__ __align__(16) __bf16 lsA[2][64 * LROW];
  __shared__ __align__(16) __bf16 lsW[2][64 * LROW];
  const int lane = threadIdx.x & 31;
  const int wv   = threadIdx.x >> 5;
  const int g    = lane >> 4;
  const int n    = lane & 15;
  const int mblk = blockIdx.x * 64;
  const int n0   = blockIdx.y * 64;
  const int nc   = K / KC;

  if (wv == 0) {
    tdm_load_2d((unsigned)(size_t)&lsA[0][0], A + (size_t)mblk * K, 64, K);
    tdm_load_2d((unsigned)(size_t)&lsW[0][0], W + (size_t)n0 * K, 64, K);
  }

  v8f acc0 = {}, acc1 = {}, acc2 = {}, acc3 = {};
  for (int c = 0; c < nc; ++c) {
    const int pb = c & 1;
    if (wv == 0) {
      if (c + 1 < nc) {
        tdm_load_2d((unsigned)(size_t)&lsA[pb ^ 1][0],
                    A + (size_t)mblk * K + (c + 1) * KC, 64, K);
        tdm_load_2d((unsigned)(size_t)&lsW[pb ^ 1][0],
                    W + (size_t)n0 * K + (c + 1) * KC, 64, K);
        __builtin_amdgcn_s_wait_tensorcnt(2);  // chunk c done (in-order)
      } else {
        __builtin_amdgcn_s_wait_tensorcnt(0);
      }
    }
    __syncthreads();  // publish chunk c to all waves

    const __bf16* la = &lsA[pb][(wv * 16 + n) * LROW];
#pragma unroll
    for (int kl = 0; kl < KC; kl += 32) {
      v16bf a = cat16(ld8(la + kl + g * 8), ld8(la + kl + g * 8 + 16));
      const __bf16* w0 = &lsW[pb][n * LROW + kl + g * 8];
      const __bf16* w1 = w0 + 16 * LROW;
      const __bf16* w2 = w0 + 32 * LROW;
      const __bf16* w3 = w0 + 48 * LROW;
      acc0 = wmma_bf16(a, cat16(ld8(w0), ld8(w0 + 16)), acc0);
      acc1 = wmma_bf16(a, cat16(ld8(w1), ld8(w1 + 16)), acc1);
      acc2 = wmma_bf16(a, cat16(ld8(w2), ld8(w2 + 16)), acc2);
      acc3 = wmma_bf16(a, cat16(ld8(w3), ld8(w3 + 16)), acc3);
    }
    __syncthreads();  // reads done before TDM overwrites this buffer
  }

  // C/D layout: vgpr r, lane-group g -> row m = r + 8*g, col = lane%16.
  float* crow = C + (size_t)(mblk + wv * 16 + 8 * g) * N + n0 + n;
#pragma unroll
  for (int r = 0; r < 8; ++r) {
    crow[(size_t)r * N +  0] = acc0[r];
    crow[(size_t)r * N + 16] = acc1[r];
    crow[(size_t)r * N + 32] = acc2[r];
    crow[(size_t)r * N + 48] = acc3[r];
  }
}

// ------------------------------ RoPE + pack --------------------------------
// raw f32 (B,S,H,D) -> bf16 (B,H,S,D) with complex rotation on (2i,2i+1).
__global__ void k_rope_pack(const float* __restrict__ raw,
                            const float* __restrict__ freqs,
                            __bf16* __restrict__ dst) {
  int i = blockIdx.x * blockDim.x + threadIdx.x;   // B*S*H*(D/2) = 2^21
  int pr = i & 31;
  int h  = (i >> 5)  & 15;
  int s  = (i >> 9)  & 2047;
  int b  = i >> 20;
  const float* q = raw + (((size_t)(b * SEQ_S + s) * NHEADS + h) << 6) + (pr << 1);
  float xr = q[0], xi = q[1];
  float fr = freqs[(((s << 5) + pr) << 1) + 0];
  float fi = freqs[(((s << 5) + pr) << 1) + 1];
  __bf16* o = dst + (((size_t)(b * NHEADS + h) * SEQ_S + s) << 6) + (pr << 1);
  o[0] = (__bf16)(xr * fr - xi * fi);
  o[1] = (__bf16)(xr * fi + xi * fr);
}

// raw f32 (B,S,H,D) -> bf16 V^T (B,H,D,S): PV B-fragment becomes two contiguous
// 8-key loads per lane (no transpose instruction needed).
__global__ void k_v_pack(const float* __restrict__ raw, __bf16* __restrict__ vt) {
  int i = blockIdx.x * blockDim.x + threadIdx.x;   // 2^22
  int dp = i & 63;
  int h  = (i >> 6)  & 15;
  int s  = (i >> 10) & 2047;
  int b  = i >> 21;
  vt[(size_t)((b * NHEADS + h) * HDIM + dp) * SEQ_S + s] = (__bf16)raw[i];
}

// ------------------------------ flash attention ----------------------------
// One wave per 16-row q-tile per (b,h); streaming softmax over 32-key steps.
__global__ __launch_bounds__(128) void k_flash_attn(
    const __bf16* __restrict__ Q, const __bf16* __restrict__ Kk,
    const __bf16* __restrict__ Vt, __bf16* __restrict__ ctx) {
  __shared__ __align__(16) __bf16 pbuf[4][16][40];  // padded rows vs bank conflicts
  const int lane = threadIdx.x & 31;
  const int wv   = threadIdx.x >> 5;
  const int g    = lane >> 4;
  const int n    = lane & 15;
  const int bh   = blockIdx.x;                 // 0..31
  const int qt   = blockIdx.y * 4 + wv;        // 0..127
  const int b    = bh >> 4, h = bh & 15;

  const __bf16* Qh = Q  + (size_t)bh * SEQ_S * HDIM;
  const __bf16* Kh = Kk + (size_t)bh * SEQ_S * HDIM;
  const __bf16* Vh = Vt + (size_t)bh * HDIM * SEQ_S;

  // Q tile 16x64 as two A-fragments (d 0..31 and 32..63), resident all loop.
  const __bf16* qrow = Qh + (size_t)(qt * 16 + n) * HDIM;
  v16bf qa0 = cat16(ld8(qrow + g * 8),      ld8(qrow + g * 8 + 16));
  v16bf qa1 = cat16(ld8(qrow + g * 8 + 32), ld8(qrow + g * 8 + 48));

  v8f acc0 = {}, acc1 = {}, acc2 = {}, acc3 = {};
  float mi[8], li[8];
#pragma unroll
  for (int r = 0; r < 8; ++r) { mi[r] = -__builtin_inff(); li[r] = 0.f; }
  const float sc = 0.125f * 1.44269504088896f;  // D^-1/2 * log2(e)

  for (int kb = 0; kb < SEQ_S; kb += 32) {
    // ---- scores: two 16x16 key tiles, each QK^T over d=64 (2 wmma) --------
    const __bf16* kr0 = Kh + (size_t)(kb + n) * HDIM + g * 8;
    const __bf16* kr1 = Kh + (size_t)(kb + 16 + n) * HDIM + g * 8;
    __builtin_prefetch(kr0 + 32 * HDIM, 0, 1);  // next kb K tile
    v8f s0v = {}, s1v = {};
    s0v = wmma_bf16(qa0, cat16(ld8(kr0),      ld8(kr0 + 16)), s0v);
    s0v = wmma_bf16(qa1, cat16(ld8(kr0 + 32), ld8(kr0 + 48)), s0v);
    s1v = wmma_bf16(qa0, cat16(ld8(kr1),      ld8(kr1 + 16)), s1v);
    s1v = wmma_bf16(qa1, cat16(ld8(kr1 + 32), ld8(kr1 + 48)), s1v);

    // ---- streaming softmax: row stats live per 16-lane group --------------
    float p0[8], p1[8], alpha[8];
#pragma unroll
    for (int r = 0; r < 8; ++r) {
      float a0 = s0v[r] * sc, a1 = s1v[r] * sc;
      float mx = fmaxf(a0, a1);
      mx = fmaxf(mx, __shfl_xor(mx, 1));
      mx = fmaxf(mx, __shfl_xor(mx, 2));
      mx = fmaxf(mx, __shfl_xor(mx, 4));
      mx = fmaxf(mx, __shfl_xor(mx, 8));
      float mnew = fmaxf(mi[r], mx);
      alpha[r] = __builtin_amdgcn_exp2f(mi[r] - mnew);
      p0[r]    = __builtin_amdgcn_exp2f(a0 - mnew);
      p1[r]    = __builtin_amdgcn_exp2f(a1 - mnew);
      float rs = p0[r] + p1[r];
      rs += __shfl_xor(rs, 1);
      rs += __shfl_xor(rs, 2);
      rs += __shfl_xor(rs, 4);
      rs += __shfl_xor(rs, 8);
      li[r] = li[r] * alpha[r] + rs;
      mi[r] = mnew;
    }

    // ---- rescale acc; transpose P (C-layout -> A-layout) through LDS ------
#pragma unroll
    for (int r = 0; r < 8; ++r) {
      acc0[r] *= alpha[r]; acc1[r] *= alpha[r];
      acc2[r] *= alpha[r]; acc3[r] *= alpha[r];
      const int m = r + 8 * g;
      pbuf[wv][m][n]      = (__bf16)p0[r];
      pbuf[wv][m][16 + n] = (__bf16)p1[r];
    }
    asm volatile("s_wait_dscnt 0" ::: "memory");   // same-wave RAW on LDS
    const __bf16* prow = &pbuf[wv][n][0];
    v16bf pa = cat16(ld8(prow + g * 8), ld8(prow + g * 8 + 16));

    // ---- PV: P(16x32) @ V(32x64); V^T rows give contiguous B-fragments ----
    const __bf16* vr0 = Vh + (size_t)n * SEQ_S + kb + g * 8;
    const __bf16* vr1 = vr0 + (size_t)16 * SEQ_S;
    const __bf16* vr2 = vr0 + (size_t)32 * SEQ_S;
    const __bf16* vr3 = vr0 + (size_t)48 * SEQ_S;
    __builtin_prefetch(vr0 + 32, 0, 1);         // next kb V tile
    acc0 = wmma_bf16(pa, cat16(ld8(vr0), ld8(vr0 + 16)), acc0);
    acc1 = wmma_bf16(pa, cat16(ld8(vr1), ld8(vr1 + 16)), acc1);
    acc2 = wmma_bf16(pa, cat16(ld8(vr2), ld8(vr2 + 16)), acc2);
    acc3 = wmma_bf16(pa, cat16(ld8(vr3), ld8(vr3 + 16)), acc3);
  }

  // ---- normalize and emit bf16 context (B,S,d) ----------------------------
  __bf16* crow = ctx + (size_t)(b * SEQ_S + qt * 16 + 8 * g) * DMODEL + h * HDIM + n;
#pragma unroll
  for (int r = 0; r < 8; ++r) {
    float inv = 1.0f / li[r];
    __bf16* o = crow + (size_t)r * DMODEL;
    o[0]  = (__bf16)(acc0[r] * inv);
    o[16] = (__bf16)(acc1[r] * inv);
    o[32] = (__bf16)(acc2[r] * inv);
    o[48] = (__bf16)(acc3[r] * inv);
  }
}

// ------------------------------- launcher ----------------------------------
extern "C" void kernel_launch(void* const* d_in, const int* in_sizes, int n_in,
                              void* d_out, int out_size, void* d_ws, size_t ws_size,
                              hipStream_t stream) {
  (void)in_sizes; (void)n_in; (void)out_size; (void)ws_size;
  const float* x     = (const float*)d_in[0];
  const float* freqs = (const float*)d_in[1];
  const float* Wq    = (const float*)d_in[2];
  const float* Wk    = (const float*)d_in[3];
  const float* Wv    = (const float*)d_in[4];
  const float* Wo    = (const float*)d_in[5];
  float* out = (float*)d_out;

  char* ws = (char*)d_ws;                       // ~58 MiB used
  __bf16* xb  = (__bf16*)(ws);                  //  8 MiB  x in bf16
  __bf16* wb  = (__bf16*)(ws + (8u  << 20));    //  2 MiB  current weight bf16
  float*  raw = (float* )(ws + (10u << 20));    // 16 MiB  raw f32 projection
  __bf16* Qb  = (__bf16*)(ws + (26u << 20));    //  8 MiB  (B,H,S,D)
  __bf16* Kb  = (__bf16*)(ws + (34u << 20));    //  8 MiB  (B,H,S,D)
  __bf16* Vt  = (__bf16*)(ws + (42u << 20));    //  8 MiB  (B,H,D,S)
  __bf16* ctx = (__bf16*)(ws + (50u << 20));    //  8 MiB  (B,S,d)

  const int M = SEQ_B * SEQ_S;   // 4096
  const int D = DMODEL;          // 1024
  dim3 gG(M / 64, D / 64), bG(128);

  k_f32_to_bf16<<<(M * D) / 256, 256, 0, stream>>>(x, xb, M * D);

  // Q projection + RoPE
  k_f32_to_bf16<<<(D * D) / 256, 256, 0, stream>>>(Wq, wb, D * D);
  k_gemm_bf16<<<gG, bG, 0, stream>>>(xb, wb, raw, M, D, D);
  k_rope_pack<<<(M * D / 2) / 256, 256, 0, stream>>>(raw, freqs, Qb);
  // K projection + RoPE
  k_f32_to_bf16<<<(D * D) / 256, 256, 0, stream>>>(Wk, wb, D * D);
  k_gemm_bf16<<<gG, bG, 0, stream>>>(xb, wb, raw, M, D, D);
  k_rope_pack<<<(M * D / 2) / 256, 256, 0, stream>>>(raw, freqs, Kb);
  // V projection (transposed pack)
  k_f32_to_bf16<<<(D * D) / 256, 256, 0, stream>>>(Wv, wb, D * D);
  k_gemm_bf16<<<gG, bG, 0, stream>>>(xb, wb, raw, M, D, D);
  k_v_pack<<<(M * D) / 256, 256, 0, stream>>>(raw, Vt);
  // flash attention -> bf16 context
  k_flash_attn<<<dim3(SEQ_B * NHEADS, SEQ_S / 64), 128, 0, stream>>>(Qb, Kb, Vt, ctx);
  // output projection straight to f32 d_out
  k_f32_to_bf16<<<(D * D) / 256, 256, 0, stream>>>(Wo, wb, D * D);
  k_gemm_bf16<<<gG, bG, 0, stream>>>(ctx, wb, out, M, D, D);
}
